// GraphSynthesizer_87187836109574
// MI455X (gfx1250) — matrix operand order repaired and checked by reading, not compiled, not run on hardware
//
#include <hip/hip_runtime.h>

typedef __attribute__((ext_vector_type(16))) __bf16        v16bf;
typedef __attribute__((ext_vector_type(8)))  float         v8f;
typedef __attribute__((ext_vector_type(4)))  unsigned int  u32x4;

#define NSYN  4096
#define CDIM  256
#define HDIM  256
#define K1    512
#define EDG   262144
#define BATCH 2
#define BM    128

// ---- workspace layout (bytes) ----
static constexpr size_t XB_OFF     = 0;                               // x bf16: 4096*256*2
static constexpr size_t W1T_OFF    = 4096ull * 256 * 2;               // W1^T bf16 [256][512]
static constexpr size_t W2T_OFF    = W1T_OFF + 512ull * 256 * 2;      // W2^T bf16 [256][256]
static constexpr size_t ACC_OFF    = W2T_OFF + 256ull * 256 * 2;      // zeroed accum region
static constexpr size_t ACC_FLOATS = 1024 + 2ull * BATCH * NSYN;      // 4 stat vecs + rowsum + colsum
static constexpr size_t BN_OFF     = ACC_OFF + ACC_FLOATS * 4;        // a1,c1,a2,c2
static constexpr size_t DINV_OFF   = BN_OFF + 1024 * 4;               // deg^-1/2 [2*4096]
static constexpr size_t Y1_OFF     = (DINV_OFF + (size_t)BATCH * NSYN * 4 + 255) & ~(size_t)255;
static constexpr size_t Y2_OFF     = Y1_OFF + (size_t)EDG * HDIM * 2;

// ---- CDNA5 async memory->LDS copy (16B per lane), tracked by ASYNCcnt ----
#if defined(__gfx1250__)
__device__ __forceinline__ void cp16_async(void* dst_lds, const void* src_global) {
    // generic LDS pointer carries the LDS byte offset in addr[31:0] (ISA 10.2)
    unsigned lds_off = (unsigned)(unsigned long long)dst_lds;
    asm volatile("global_load_async_to_lds_b128 %0, %1, off"
                 :: "v"(lds_off), "v"(src_global) : "memory");
}
#define ASYNC_WAIT(N) asm volatile("s_wait_asynccnt " #N ::: "memory")
#else
__device__ __forceinline__ void cp16_async(void* dst_lds, const void* src_global) {
    *(u32x4*)dst_lds = *(const u32x4*)src_global;
}
#define ASYNC_WAIT(N)
#endif

// ---------------- conversion kernels ----------------
__global__ void k_cvt_x(const float* __restrict__ x, __bf16* __restrict__ xb, int n) {
    int i = blockIdx.x * 256 + threadIdx.x;
    if (i < n) xb[i] = (__bf16)x[i];
}

// wt[n*K + k] = w[k*N + n]   (pre-transpose so GEMM B-tiles stage contiguously)
__global__ void k_cvt_wT(const float* __restrict__ w, __bf16* __restrict__ wt, int K, int N) {
    int i = blockIdx.x * 256 + threadIdx.x;
    if (i < K * N) {
        int n = i / K, k = i - n * K;
        wt[(size_t)n * K + k] = (__bf16)w[(size_t)k * N + n];
    }
}

// ---------------- GEMM1: gather + [E,512]x[512,256] + bias, raw bf16 out + BN stats ----------
// Double-buffered: tile kt+1 streams memory->LDS via async copies while WMMAs consume tile kt.
__global__ __launch_bounds__(256, 1) void k_gemm1(
    const __bf16* __restrict__ xb, const __bf16* __restrict__ w1t,
    const float* __restrict__ bias, const int* __restrict__ rows,
    const int* __restrict__ cols, __bf16* __restrict__ y1,
    float* __restrict__ gsum, float* __restrict__ gsqs)
{
    __shared__ __align__(16) __bf16 lA[2][BM][32];
    __shared__ __align__(16) __bf16 lB[2][256][32];
    __shared__ int   lR[BM], lC[BM];
    __shared__ float lSum[256], lSqs[256];

    const int tid = threadIdx.x, lane = tid & 31, wave = tid >> 5;
    const int e0 = blockIdx.x * BM;

    if (tid < BM) { int e = e0 + tid; lR[tid] = rows[e] * CDIM; lC[tid] = cols[e] * CDIM; }
    lSum[tid] = 0.f; lSqs[tid] = 0.f;

    v8f acc[16];
    v8f zero = {};
    #pragma unroll
    for (int i = 0; i < 16; ++i) acc[i] = zero;
    __syncthreads();   // lR/lC visible before first stage

    const int m    = wave * 16 + (lane & 15);
    const int half = lane >> 4;
    // A chunks: rows tid>>2 and (tid+256)>>2, 16B sub-chunk tid&3
    const int ra0 = tid >> 2, ra1 = (tid + 256) >> 2, sa = tid & 3;
    // B chunks: N rows tid>>2 + i*64, 16B sub-chunk tid&3
    const int nb = tid >> 2, sb = tid & 3;

    // issue one K-tile of async copies (6 per wave: 2 A + 4 B)
    auto stage = [&](int kt, int buf) {
        const int k0 = kt * 32;
        const int off0 = (k0 < CDIM) ? (lR[ra0] + k0) : (lC[ra0] + (k0 - CDIM));
        const int off1 = (k0 < CDIM) ? (lR[ra1] + k0) : (lC[ra1] + (k0 - CDIM));
        cp16_async(&lA[buf][ra0][sa * 8], &xb[off0 + sa * 8]);
        cp16_async(&lA[buf][ra1][sa * 8], &xb[off1 + sa * 8]);
        #pragma unroll
        for (int i = 0; i < 4; ++i)
            cp16_async(&lB[buf][nb + i * 64][sb * 8],
                       &w1t[(size_t)(nb + i * 64) * K1 + k0 + sb * 8]);
    };

    stage(0, 0);
    for (int kt = 0; kt < 16; ++kt) {
        const int cur = kt & 1;
        if (kt < 15) {
            stage(kt + 1, cur ^ 1);   // prefetch next tile into other buffer
            ASYNC_WAIT(6);            // retire tile kt's 6 copies (in-order completion)
        } else {
            ASYNC_WAIT(0);
        }
        __syncthreads();

        v16bf a;
        ((u32x4*)&a)[0] = *(const u32x4*)&lA[cur][m][half * 8];
        ((u32x4*)&a)[1] = *(const u32x4*)&lA[cur][m][16 + half * 8];
        #pragma unroll
        for (int nt = 0; nt < 16; ++nt) {
            v16bf b;
            const __bf16* bp = &lB[cur][nt * 16 + (lane & 15)][half * 8];
            ((u32x4*)&b)[0] = ((const u32x4*)bp)[0];
            ((u32x4*)&b)[1] = *(const u32x4*)(bp + 16);
            acc[nt] = __builtin_amdgcn_wmma_f32_16x16x32_bf16(
                false, a, false, b, (short)0, acc[nt], false, false);
        }
        __syncthreads();   // all reads of buf[cur] done before it is refilled
    }

    #pragma unroll
    for (int nt = 0; nt < 16; ++nt) {
        const int n = nt * 16 + (lane & 15);
        const float bv = bias[n];
        float s = 0.f, s2 = 0.f;
        #pragma unroll
        for (int j = 0; j < 8; ++j) {
            const int mm = wave * 16 + j + half * 8;
            float y = acc[nt][j] + bv;
            y1[(size_t)(e0 + mm) * HDIM + n] = (__bf16)y;
            s += y; s2 += y * y;
        }
        atomicAdd(&lSum[n], s);
        atomicAdd(&lSqs[n], s2);
    }
    __syncthreads();
    atomicAdd(&gsum[tid], lSum[tid]);
    atomicAdd(&gsqs[tid], lSqs[tid]);
}

// ---------------- BN finalize: a = g*rsqrt(var+eps), c = be - mean*a ----------------
__global__ void k_bnfin(const float* __restrict__ gsum, const float* __restrict__ gsqs,
                        const float* __restrict__ g, const float* __restrict__ be,
                        float* __restrict__ a, float* __restrict__ c)
{
    int i = threadIdx.x;
    float mean = gsum[i] * (1.0f / EDG);
    float var  = gsqs[i] * (1.0f / EDG) - mean * mean;
    float av   = g[i] * rsqrtf(var + 1e-5f);
    a[i] = av;
    c[i] = be[i] - mean * av;
}

// ---------------- GEMM2: relu(bn1(Y1)) x [256,256] + bias, raw bf16 out + BN2 stats ----------
// B tile streams async into LDS, overlapped with the VALU work of the fused BN1+ReLU A staging.
__global__ __launch_bounds__(256, 1) void k_gemm2(
    const __bf16* __restrict__ y1, const __bf16* __restrict__ w2t,
    const float* __restrict__ bias, const float* __restrict__ bnA,
    const float* __restrict__ bnC, __bf16* __restrict__ y2,
    float* __restrict__ gsum, float* __restrict__ gsqs)
{
    __shared__ __align__(16) __bf16 lA[BM][32];
    __shared__ __align__(16) __bf16 lB[256][32];
    __shared__ float lbnA[256], lbnC[256];
    __shared__ float lSum[256], lSqs[256];

    const int tid = threadIdx.x, lane = tid & 31, wave = tid >> 5;
    const int e0 = blockIdx.x * BM;

    lbnA[tid] = bnA[tid]; lbnC[tid] = bnC[tid];
    lSum[tid] = 0.f; lSqs[tid] = 0.f;

    v8f acc[16];
    v8f zero = {};
    #pragma unroll
    for (int i = 0; i < 16; ++i) acc[i] = zero;
    __syncthreads();

    const int m    = wave * 16 + (lane & 15);
    const int half = lane >> 4;
    const int nb = tid >> 2, sb = tid & 3;

    for (int kt = 0; kt < 8; ++kt) {
        const int k0 = kt * 32;
        // async B-tile copies first: they fly while we convert A below
        #pragma unroll
        for (int i = 0; i < 4; ++i)
            cp16_async(&lB[nb + i * 64][sb * 8],
                       &w2t[(size_t)(nb + i * 64) * HDIM + k0 + sb * 8]);
        // stage A with fused BN1 + ReLU (must pass through VGPRs)
        #pragma unroll
        for (int i = 0; i < 2; ++i) {
            int q = tid + i * 256, r = q >> 2, sub = q & 3;
            alignas(16) __bf16 t[8];
            *(u32x4*)t = *(const u32x4*)&y1[(size_t)(e0 + r) * HDIM + k0 + sub * 8];
            #pragma unroll
            for (int j = 0; j < 8; ++j) {
                int k = k0 + sub * 8 + j;
                float f = fmaxf(fmaf(lbnA[k], (float)t[j], lbnC[k]), 0.f);
                t[j] = (__bf16)f;
            }
            *(u32x4*)&lA[r][sub * 8] = *(u32x4*)t;
        }
        if (kt + 1 < 8) __builtin_prefetch(&y1[(size_t)(e0 + (tid >> 2)) * HDIM + k0 + 32], 0, 0);
        ASYNC_WAIT(0);
        __syncthreads();

        v16bf a;
        ((u32x4*)&a)[0] = *(const u32x4*)&lA[m][half * 8];
        ((u32x4*)&a)[1] = *(const u32x4*)&lA[m][16 + half * 8];
        #pragma unroll
        for (int nt = 0; nt < 16; ++nt) {
            v16bf b;
            const __bf16* bp = &lB[nt * 16 + (lane & 15)][half * 8];
            ((u32x4*)&b)[0] = ((const u32x4*)bp)[0];
            ((u32x4*)&b)[1] = *(const u32x4*)(bp + 16);
            acc[nt] = __builtin_amdgcn_wmma_f32_16x16x32_bf16(
                false, a, false, b, (short)0, acc[nt], false, false);
        }
        __syncthreads();
    }

    #pragma unroll
    for (int nt = 0; nt < 16; ++nt) {
        const int n = nt * 16 + (lane & 15);
        const float bv = bias[n];
        float s = 0.f, s2 = 0.f;
        #pragma unroll
        for (int j = 0; j < 8; ++j) {
            const int mm = wave * 16 + j + half * 8;
            float y = acc[nt][j] + bv;
            y2[(size_t)(e0 + mm) * HDIM + n] = (__bf16)y;
            s += y; s2 += y * y;
        }
        atomicAdd(&lSum[n], s);
        atomicAdd(&lSqs[n], s2);
    }
    __syncthreads();
    atomicAdd(&gsum[tid], lSum[tid]);
    atomicAdd(&gsqs[tid], lSqs[tid]);
}

// ---------------- edge head: sigmoid(relu(bn2(Y2)) . W3 + b3), scatter + degree atomics ----------
__global__ __launch_bounds__(256) void k_edge(
    const __bf16* __restrict__ y2, const float* __restrict__ bnA,
    const float* __restrict__ bnC, const float* __restrict__ w3,
    const float* __restrict__ b3, const int* __restrict__ rows,
    const int* __restrict__ cols, const int* __restrict__ bidx,
    float* __restrict__ adj, float* __restrict__ rsum, float* __restrict__ csum)
{
    const int lane = threadIdx.x & 31;
    const int e    = blockIdx.x * 8 + (threadIdx.x >> 5);
    const int k0   = lane * 8;

    alignas(16) __bf16 t[8];
    *(u32x4*)t = *(const u32x4*)&y2[(size_t)e * HDIM + k0];
    float p = 0.f;
    #pragma unroll
    for (int j = 0; j < 8; ++j) {
        int k = k0 + j;
        float f = fmaxf(fmaf(bnA[k], (float)t[j], bnC[k]), 0.f);
        p = fmaf(f, w3[k], p);
    }
    #pragma unroll
    for (int off = 16; off > 0; off >>= 1) p += __shfl_xor(p, off, 32);

    if (lane == 0) {
        float s = 1.0f / (1.0f + __expf(-(p + b3[0])));
        int b = bidx[e], r = rows[e], c = cols[e];
        adj[((size_t)b << 24) | ((size_t)r << 12) | (size_t)c] = s;
        atomicAdd(&rsum[b * NSYN + r], s);
        atomicAdd(&csum[b * NSYN + c], s);
    }
}

// ---------------- deg^-1/2 (deg = 1 + 0.5*(rowsum + colsum), always >= 1) ----------------
__global__ void k_dinv(const float* __restrict__ rsum, const float* __restrict__ csum,
                       float* __restrict__ dinv)
{
    int i = blockIdx.x * 256 + threadIdx.x;
    if (i < BATCH * NSYN) dinv[i] = rsqrtf(1.0f + 0.5f * (rsum[i] + csum[i]));
}

// ---------------- in-place symmetrize + add I + D^-1/2 A D^-1/2, tile pairs ----------------
__global__ __launch_bounds__(256) void k_sym(float* __restrict__ adj, const float* __restrict__ dinv)
{
    const int ti = blockIdx.x, tj = blockIdx.y, b = blockIdx.z;
    if (ti > tj) return;
    __shared__ float tB[32][33];
    __shared__ float tT[32][33];
    const int tx = threadIdx.x & 31, ty = threadIdx.x >> 5;
    const size_t base = (size_t)b << 24;
    const int i0 = ti * 32, j0 = tj * 32;

    #pragma unroll
    for (int r = 0; r < 4; ++r) {
        int y = ty + r * 8;
        tB[y][tx] = adj[base + (size_t)(j0 + y) * NSYN + i0 + tx];
    }
    __syncthreads();
    #pragma unroll
    for (int r = 0; r < 4; ++r) {
        int y = ty + r * 8;
        int i = i0 + y, j = j0 + tx;
        float a = adj[base + (size_t)i * NSYN + j];
        float v = 0.5f * (a + tB[tx][y]);
        if (i == j) v += 1.0f;
        v *= dinv[b * NSYN + i] * dinv[b * NSYN + j];
        adj[base + (size_t)i * NSYN + j] = v;
        tT[tx][y] = v;                     // value for output position (j, i)
    }
    __syncthreads();
    #pragma unroll
    for (int r = 0; r < 4; ++r) {
        int y = ty + r * 8;
        adj[base + (size_t)(j0 + y) * NSYN + i0 + tx] = tT[y][tx];
    }
}

// ---------------- host launcher ----------------
extern "C" void kernel_launch(void* const* d_in, const int* in_sizes, int n_in,
                              void* d_out, int out_size, void* d_ws, size_t ws_size,
                              hipStream_t stream)
{
    (void)in_sizes; (void)n_in; (void)ws_size;
    const float* x   = (const float*)d_in[0];
    const float* W1  = (const float*)d_in[1];
    const float* b1  = (const float*)d_in[2];
    const float* g1  = (const float*)d_in[3];
    const float* be1 = (const float*)d_in[4];
    const float* W2  = (const float*)d_in[5];
    const float* b2  = (const float*)d_in[6];
    const float* g2  = (const float*)d_in[7];
    const float* be2 = (const float*)d_in[8];
    const float* W3  = (const float*)d_in[9];
    const float* b3  = (const float*)d_in[10];
    const int* rows  = (const int*)d_in[11];
    const int* cols  = (const int*)d_in[12];
    const int* bidx  = (const int*)d_in[13];

    char* ws = (char*)d_ws;
    __bf16* xb   = (__bf16*)(ws + XB_OFF);
    __bf16* w1t  = (__bf16*)(ws + W1T_OFF);
    __bf16* w2t  = (__bf16*)(ws + W2T_OFF);
    float*  accr = (float*)(ws + ACC_OFF);
    float *gsum1 = accr, *gsqs1 = accr + 256, *gsum2 = accr + 512, *gsqs2 = accr + 768;
    float*  rsum = accr + 1024;
    float*  csum = rsum + BATCH * NSYN;
    float*  bn   = (float*)(ws + BN_OFF);
    float *a1 = bn, *c1 = bn + 256, *a2 = bn + 512, *c2 = bn + 768;
    float*  dinv = (float*)(ws + DINV_OFF);
    __bf16* y1   = (__bf16*)(ws + Y1_OFF);
    __bf16* y2   = (__bf16*)(ws + Y2_OFF);
    float*  adj  = (float*)d_out;

    hipMemsetAsync(d_out, 0, (size_t)out_size * sizeof(float), stream);
    hipMemsetAsync(accr, 0, ACC_FLOATS * sizeof(float), stream);

    k_cvt_x <<<(NSYN * CDIM + 255) / 256, 256, 0, stream>>>(x, xb, NSYN * CDIM);
    k_cvt_wT<<<(K1 * HDIM + 255) / 256, 256, 0, stream>>>(W1, w1t, K1, HDIM);
    k_cvt_wT<<<(HDIM * HDIM + 255) / 256, 256, 0, stream>>>(W2, w2t, HDIM, HDIM);

    k_gemm1<<<EDG / BM, 256, 0, stream>>>(xb, w1t, b1, rows, cols, y1, gsum1, gsqs1);
    k_bnfin<<<1, 256, 0, stream>>>(gsum1, gsqs1, g1, be1, a1, c1);
    k_gemm2<<<EDG / BM, 256, 0, stream>>>(y1, w2t, b2, a1, c1, y2, gsum2, gsqs2);
    k_bnfin<<<1, 256, 0, stream>>>(gsum2, gsqs2, g2, be2, a2, c2);

    k_edge<<<EDG / 8, 256, 0, stream>>>(y2, a2, c2, W3, b3, rows, cols, bidx, adj, rsum, csum);
    k_dinv<<<(BATCH * NSYN + 255) / 256, 256, 0, stream>>>(rsum, csum, dinv);

    dim3 gs(NSYN / 32, NSYN / 32, BATCH);
    k_sym<<<gs, 256, 0, stream>>>(adj, dinv);
}
